// STFT_83124797047169
// MI455X (gfx1250) — matrix operand order repaired and checked by reading, not compiled, not run on hardware
//
#include <hip/hip_runtime.h>
#include <hip/hip_bf16.h>
#include <math.h>

typedef __attribute__((ext_vector_type(16))) _Float16 v16h;
typedef __attribute__((ext_vector_type(8)))  _Float16 v8h;
typedef __attribute__((ext_vector_type(8)))  float    v8f;

#define WIN     1024
#define HOP     256
#define NDIM    513              // N_FFT/2 + 1
#define NCH     1026             // 2*NDIM conv output channels
#define BATCH   32
#define LSIG    160000
#define LPAD    (LSIG + WIN)     // 161024 reflect-padded length
#define FRAMES  626
#define NTOT    (BATCH * FRAMES) // 20032 total frames == GEMM N
#define NTILES  (NTOT / 16)      // 1252 (exact)
#define MTILES  33               // ceil(513/16)
#define WAVES_PER_BLOCK 8

// ---- prologue 1: weight f32 -> f16 (2.1 MB, L2-resident afterwards) ----
__global__ void cvt_weight(const float* __restrict__ w, _Float16* __restrict__ wh, int n) {
  int i = blockIdx.x * blockDim.x + threadIdx.x;
  if (i < n) wh[i] = (_Float16)w[i];
}

// ---- prologue 2: reflect-pad x and convert to f16 (10.3 MB) ----
__global__ void pad_x(const float* __restrict__ x, _Float16* __restrict__ xh) {
  int i = blockIdx.x * blockDim.x + threadIdx.x;
  if (i >= BATCH * LPAD) return;
  int b = i / LPAD;
  int p = i % LPAD;
  int o = p - (WIN / 2);
  if (o < 0) o = -o;                       // reflect left
  if (o >= LSIG) o = 2 * LSIG - 2 - o;     // reflect right
  xh[i] = (_Float16)x[b * LSIG + o];
}

// ---- fused STFT GEMM: two v_wmma per K-step (real + imag), mags/phase epilogue ----
__global__ void __launch_bounds__(WAVES_PER_BLOCK * 32)
stft_wmma(const _Float16* __restrict__ Wh,   // 1026 x 1024, row-major
          const _Float16* __restrict__ Xh,   // 32 x 161024 padded signal
          float* __restrict__ out) {         // [mags | phase], each 32*513*626
  const int lane  = threadIdx.x & 31;
  const int wave  = threadIdx.x >> 5;
  const int nTile = blockIdx.x * WAVES_PER_BLOCK + wave;
  const int mTile = blockIdx.y;
  if (nTile >= NTILES) return;               // wave-uniform: EXEC stays all-ones

  const int hi = lane >> 4;                  // lane half: selects K sub-chunk
  const int lo = lane & 15;

  // A-fragment row for this lane (ISA: lanes 0-15 and 16-31 both hold M = lane&15)
  int cRe = mTile * 16 + lo;
  int rowRe = cRe < (NDIM - 1) ? cRe : (NDIM - 1);  // clamp partial tile to row 512
  int rowIm = rowRe + NDIM;                          // <= 1025, always in-bounds

  // B-fragment column = frame index for this lane; frame data is K-contiguous
  const int n  = nTile * 16 + lo;            // < 20032 always
  const int bb = n / FRAMES;
  const int tt = n % FRAMES;

  const _Float16* bbase  = Xh + (size_t)bb * LPAD + (size_t)tt * HOP + hi * 8;
  const _Float16* arbase = Wh + (size_t)rowRe * WIN + hi * 8;
  const _Float16* aibase = Wh + (size_t)rowIm * WIN + hi * 8;

  v8f accRe = {};
  v8f accIm = {};

#pragma unroll 4
  for (int k = 0; k < WIN; k += 32) {
    // A fragment: halfs 0..7 = K (k + 8*hi + 0..7), halfs 8..15 = +16
    v8h arl = *(const v8h*)(arbase + k);
    v8h arh = *(const v8h*)(arbase + k + 16);
    v8h ail = *(const v8h*)(aibase + k);
    v8h aih = *(const v8h*)(aibase + k + 16);
    // B fragment: halfs 0..7 = K (k + 8*hi + 0..7), halfs 8..15 = +16
    v8h bl  = *(const v8h*)(bbase + k);
    v8h bh  = *(const v8h*)(bbase + k + 16);

    v16h aRe, aIm, bF;
#pragma unroll
    for (int i = 0; i < 8; ++i) {
      aRe[i] = arl[i]; aRe[i + 8] = arh[i];
      aIm[i] = ail[i]; aIm[i + 8] = aih[i];
      bF[i]  = bl[i];  bF[i + 8]  = bh[i];
    }

    accRe = __builtin_amdgcn_wmma_f32_16x16x32_f16(
        false, aRe, false, bF, (short)0, accRe, false, false);
    accIm = __builtin_amdgcn_wmma_f32_16x16x32_f16(
        false, aIm, false, bF, (short)0, accIm, false, false);
  }

  // Epilogue: C layout -> output. VGPR r, lane half hi: M = r + 8*hi, N = lo.
  const size_t halfOut = (size_t)BATCH * NDIM * FRAMES;  // 10,276,416
#pragma unroll
  for (int r = 0; r < 8; ++r) {
    int c = mTile * 16 + hi * 8 + r;
    if (c < NDIM) {
      float re = accRe[r];
      float im = accIm[r];
      float mag = sqrtf(re * re + im * im);
      float ph  = atan2f(im, re);
      size_t idx = ((size_t)bb * NDIM + c) * FRAMES + tt;
      out[idx]           = mag;
      out[halfOut + idx] = ph;
    }
  }
}

extern "C" void kernel_launch(void* const* d_in, const int* in_sizes, int n_in,
                              void* d_out, int out_size, void* d_ws, size_t ws_size,
                              hipStream_t stream) {
  const float* x = (const float*)d_in[0];   // (32,1,160000) f32
  const float* w = (const float*)d_in[1];   // (1026,1,1024) f32
  float* out = (float*)d_out;               // mags ++ phase

  const int wElems = NCH * WIN;             // 1,050,624
  _Float16* wh = (_Float16*)d_ws;
  _Float16* xh = (_Float16*)((char*)d_ws + (size_t)wElems * sizeof(_Float16)); // +2,101,248 B

  cvt_weight<<<(wElems + 255) / 256, 256, 0, stream>>>(w, wh, wElems);
  pad_x<<<(BATCH * LPAD + 255) / 256, 256, 0, stream>>>(x, xh);

  dim3 grid((NTILES + WAVES_PER_BLOCK - 1) / WAVES_PER_BLOCK, MTILES);
  stft_wmma<<<grid, WAVES_PER_BLOCK * 32, 0, stream>>>(wh, xh, out);
}
